// ChatGPT_8340826489267
// MI455X (gfx1250) — compile-verified, matrix operand order checked by reading, not run on hardware
//
#include <hip/hip_runtime.h>
#include <hip/hip_bf16.h>
#include <math.h>

// Problem constants (match reference)
#define BB 2
#define TT 1024
#define CC 1024
#define HH 16
#define HD 64
#define LLAYERS 3
#define VV 32000
#define MM (BB*TT)          // 2048 rows of activations

typedef __bf16 bf16;
typedef __attribute__((ext_vector_type(16))) __bf16 v16bf;
typedef __attribute__((ext_vector_type(8)))  __bf16 v8bf;
typedef __attribute__((ext_vector_type(8)))  float  v8f;

// ---------------------------------------------------------------------------
// CDNA5 async global->LDS copy (16B per lane), tracked with ASYNCcnt.
// VDST = per-lane LDS byte address, VADDR = 64-bit global address.
// Generic pointers to LDS carry the LDS offset in their low 32 bits.
// ---------------------------------------------------------------------------
static __device__ inline void async_copy16(void* lds_ptr, const void* gptr) {
  unsigned lds = (unsigned)(unsigned long long)lds_ptr;
  unsigned long long ga = (unsigned long long)gptr;
  asm volatile("global_load_async_to_lds_b128 %0, %1, off"
               :: "v"(lds), "v"(ga) : "memory");
}
static __device__ inline void wait_async0() {
  asm volatile("s_wait_asynccnt 0" ::: "memory");
}

// ---------------------------------------------------------------------------
// WMMA helpers (gfx1250: V_WMMA_F32_16X16X32_BF16, wave32)
// A fragment (16x32 bf16): lane m = lane&15; element i -> K = kbase +
//   ((lane&16)?8:0) + (i<8 ? i : 8+i).  Two contiguous 16B LDS reads.
// B fragment (32x16 bf16): same K pattern, lane holds column n = lane&15.
// C/D fragment (16x16 f32): element r -> row r + ((lane&16)?8:0), col lane&15.
// ---------------------------------------------------------------------------
static __device__ inline v16bf ld_frag(const bf16* base, int ld, int row,
                                       int kbase, int lane) {
  const bf16* p = base + row * ld + kbase + ((lane & 16) ? 8 : 0);
  v8bf lo = *(const v8bf*)(p);
  v8bf hi = *(const v8bf*)(p + 16);
  return __builtin_shufflevector(lo, hi, 0,1,2,3,4,5,6,7,8,9,10,11,12,13,14,15);
}

static __device__ inline v8f wmma_bf16(v16bf a, v16bf b, v8f c) {
  return __builtin_amdgcn_wmma_f32_16x16x32_bf16(
      /*neg_a=*/false, a, /*neg_b=*/false, b,
      /*c_mod=*/(short)0, c, /*reuse_a=*/false, /*reuse_b=*/false);
}

// ---------------------------------------------------------------------------
// Embedding: h[b*T+t, c] = W_emb[x[b,t], c] + W_pos[t, c]
// ---------------------------------------------------------------------------
__global__ __launch_bounds__(256) void embed_kernel(
    const int* __restrict__ x, const float* __restrict__ Wemb,
    const float* __restrict__ Wpos, float* __restrict__ h) {
  int row = blockIdx.x;
  int t = row % TT;
  int tok = x[row];
  const float* we = Wemb + (size_t)tok * CC;
  const float* wp = Wpos + (size_t)t * CC;
  float* hr = h + (size_t)row * CC;
  for (int c = threadIdx.x; c < CC; c += blockDim.x) hr[c] = we[c] + wp[c];
}

// ---------------------------------------------------------------------------
// LayerNorm over C=1024, output bf16 (feeds WMMA A operands)
// ---------------------------------------------------------------------------
__global__ __launch_bounds__(256) void ln_kernel(
    const float* __restrict__ x, const float* __restrict__ g,
    const float* __restrict__ b, bf16* __restrict__ out) {
  __shared__ float red[64];
  int row = blockIdx.x;
  const float* xr = x + (size_t)row * CC;
  float s = 0.f, ss = 0.f;
  for (int c = threadIdx.x; c < CC; c += blockDim.x) {
    float v = xr[c]; s += v; ss += v * v;
  }
  for (int off = 16; off > 0; off >>= 1) {
    s  += __shfl_xor(s,  off, 32);
    ss += __shfl_xor(ss, off, 32);
  }
  int wid = threadIdx.x >> 5, lane = threadIdx.x & 31;
  if (lane == 0) { red[wid] = s; red[wid + 32] = ss; }
  __syncthreads();
  if (threadIdx.x == 0) {
    float ts = 0.f, tss = 0.f;
    for (int i = 0; i < 8; ++i) { ts += red[i]; tss += red[i + 32]; }
    float mean = ts / CC;
    float var  = tss / CC - mean * mean;
    red[0] = mean;
    red[1] = rsqrtf(var + 1e-5f);
  }
  __syncthreads();
  float mean = red[0], inv = red[1];
  bf16* orow = out + (size_t)row * CC;
  for (int c = threadIdx.x; c < CC; c += blockDim.x)
    orow[c] = (bf16)((xr[c] - mean) * inv * g[c] + b[c]);
}

// ---------------------------------------------------------------------------
// GEMM: OUT[M,N] = epilogue(A[M,K](bf16) @ B[K,N](fp32->bf16) + bias)
//  - bT=1: B element (k,n) = Bsrc[n*K + k]  (tied-embedding logits)
//  - doGelu: exact gelu;  resid: fp32 residual added;  outF xor outB dest.
// Block: 256 thr (8 waves). Macro tile 128x64, K-tile 64, double-buffered LDS,
// A tile staged with GLOBAL_LOAD_ASYNC_TO_LDS_B128 (ASYNCcnt pipeline).
// ---------------------------------------------------------------------------
#define LDA 72   // padded row length (144B, 16B aligned)

__global__ __launch_bounds__(256) void gemm_kernel(
    const bf16*  __restrict__ A, const float* __restrict__ Bsrc,
    const float* __restrict__ bias, const float* __restrict__ resid,
    float* __restrict__ outF, bf16* __restrict__ outB,
    int N, int K, int bT, int doGelu) {
  __shared__ bf16 As[2][128 * LDA];
  __shared__ bf16 Bs[2][64 * LDA];

  const int tid = threadIdx.x, lane = tid & 31, wid = tid >> 5;
  const int m0 = blockIdx.y * 128, n0 = blockIdx.x * 64;
  const int wm = (wid >> 1) * 32, wn = (wid & 1) * 32;

  const v8f vzero = {0.f,0.f,0.f,0.f,0.f,0.f,0.f,0.f};
  v8f acc[2][2];
  acc[0][0] = vzero; acc[0][1] = vzero; acc[1][0] = vzero; acc[1][1] = vzero;

  // Stage one 128x64 A tile (async DMA) + 64x64 B tile (fp32->bf16) into buf.
  auto stage = [&](int k0, int buf) {
    // A: 128 rows x 64 bf16 = 1024 x 16B chunks, 4 per thread, async to LDS.
#pragma unroll
    for (int i = 0; i < 4; ++i) {
      int idx = tid + i * 256;            // 0..1023
      int row = idx >> 3, q = idx & 7;    // 8 chunks per row
      async_copy16(&As[buf][row * LDA + q * 8],
                   A + (size_t)(m0 + row) * K + k0 + q * 8);
    }
    // B: 64x64 fp32 -> bf16, batched float4 loads then packed stores.
    if (bT) {
      // Bsrc row n holds K contiguous: read along k, store contiguous in Bs.
#pragma unroll
      for (int i = 0; i < 4; ++i) {
        int idx = tid + i * 256;          // 0..1023
        int n = idx >> 4, kq = idx & 15;  // k = kq*4
        float4 f = *(const float4*)&Bsrc[(size_t)(n0 + n) * K + k0 + kq * 4];
        union { bf16 h[4]; uint2 u; } pk;
        pk.h[0] = (bf16)f.x; pk.h[1] = (bf16)f.y;
        pk.h[2] = (bf16)f.z; pk.h[3] = (bf16)f.w;
        *(uint2*)&Bs[buf][n * LDA + kq * 4] = pk.u;
      }
    } else {
      // Bsrc row k holds N contiguous: read along n, scatter into Bs[n][k].
#pragma unroll
      for (int i = 0; i < 4; ++i) {
        int idx = tid + i * 256;
        int k = idx >> 4, nq = idx & 15;  // n = nq*4
        float4 f = *(const float4*)&Bsrc[(size_t)(k0 + k) * N + n0 + nq * 4];
        Bs[buf][(nq * 4 + 0) * LDA + k] = (bf16)f.x;
        Bs[buf][(nq * 4 + 1) * LDA + k] = (bf16)f.y;
        Bs[buf][(nq * 4 + 2) * LDA + k] = (bf16)f.z;
        Bs[buf][(nq * 4 + 3) * LDA + k] = (bf16)f.w;
      }
    }
  };

  const int nk = K >> 6;                  // K / 64
  stage(0, 0);
  for (int it = 0; it < nk; ++it) {
    int cur = it & 1;
    wait_async0();                        // own async copies landed in LDS
    __syncthreads();                      // everyone's tile (A async + B ds) visible
    if (it + 1 < nk) {
      stage((it + 1) << 6, cur ^ 1);      // overlap next tile DMA with WMMA
      __builtin_prefetch(bT ? &Bsrc[(size_t)n0 * K + ((it + 1) << 6)]
                            : &Bsrc[(size_t)((it + 1) << 6) * N + n0], 0, 2);
    }
#pragma unroll
    for (int ks = 0; ks < 64; ks += 32) {
      v16bf bfr[2];
#pragma unroll
      for (int nt = 0; nt < 2; ++nt)
        bfr[nt] = ld_frag(&Bs[cur][0], LDA, wn + nt * 16 + (lane & 15), ks, lane);
#pragma unroll
      for (int mt = 0; mt < 2; ++mt) {
        v16bf af = ld_frag(&As[cur][0], LDA, wm + mt * 16 + (lane & 15), ks, lane);
#pragma unroll
        for (int nt = 0; nt < 2; ++nt)
          acc[mt][nt] = wmma_bf16(af, bfr[nt], acc[mt][nt]);
      }
    }
  }

  // --- epilogue: bias, exact GELU, residual, fp32 or bf16 store ---
#pragma unroll
  for (int mt = 0; mt < 2; ++mt)
#pragma unroll
    for (int nt = 0; nt < 2; ++nt) {
      int gcol = n0 + wn + nt * 16 + (lane & 15);
      float bv = bias ? bias[gcol] : 0.f;
#pragma unroll
      for (int r = 0; r < 8; ++r) {
        int grow = m0 + wm + mt * 16 + r + ((lane & 16) ? 8 : 0);
        float v = acc[mt][nt][r] + bv;
        if (doGelu) v = 0.5f * v * (1.f + erff(v * 0.70710678118654752f));
        size_t idx = (size_t)grow * N + gcol;
        if (resid) v += resid[idx];
        if (outF) outF[idx] = v; else outB[idx] = (bf16)v;
      }
    }
}

// ---------------------------------------------------------------------------
// Flash attention: grid (T/64, B*H), block 128 (4 waves x 16 query rows).
// qkv is bf16 laid out [B,T,3,H,HD]; output attnb bf16 [B*T, C].
// Q/K blocks staged with async global->LDS; V transposed through registers.
// S = Q K^T * scale (causal), online softmax, O += P V.  All MMs are WMMA.
// ---------------------------------------------------------------------------
__global__ __launch_bounds__(128) void attn_kernel(
    const bf16* __restrict__ qkv, bf16* __restrict__ attnb) {
  constexpr int LD = 72;                         // 144B rows (16B-aligned)
  __shared__ bf16 Qs[64 * LD];                   // Q block, [row][d]
  __shared__ bf16 Ks[64 * LD];                   // K block, [key][d]
  __shared__ bf16 Vt[64 * LD];                   // V block transposed [d][key]
  __shared__ bf16 Ps[64 * LD];                   // probabilities [row][key]

  const int qb = blockIdx.x;
  const int bb = blockIdx.y / HH, hh = blockIdx.y % HH;
  const int tid = threadIdx.x, lane = tid & 31, wid = tid >> 5;
  const int q0 = qb * 64;
  const int lrow = tid >> 1, lhalf = (tid & 1) * 32;   // staging coords

  // Q block: async DMA straight into LDS (bf16 -> bf16)
  {
    const bf16* g = qkv + ((size_t)(bb * TT + q0 + lrow) * 3 * CC) + hh * HD + lhalf;
#pragma unroll
    for (int j = 0; j < 4; ++j)
      async_copy16(&Qs[lrow * LD + lhalf + j * 8], g + j * 8);
  }

  float m_i[8], l_i[8];
  v8f o[4];
  const v8f vzero = {0.f,0.f,0.f,0.f,0.f,0.f,0.f,0.f};
#pragma unroll
  for (int r = 0; r < 8; ++r) { m_i[r] = -1e30f; l_i[r] = 0.f; }
#pragma unroll
  for (int d = 0; d < 4; ++d) o[d] = vzero;

  const int colb0 = lane & 15;
  const int rowbase = q0 + wid * 16 + ((lane & 16) ? 8 : 0);

  for (int kb = 0; kb <= qb; ++kb) {
    __syncthreads();   // previous iteration's readers of Ks/Vt are done
    {
      size_t base = ((size_t)(bb * TT + kb * 64 + lrow) * 3 * CC) + hh * HD + lhalf;
      // K block: async DMA
      const bf16* gk = qkv + base + CC;
#pragma unroll
      for (int j = 0; j < 4; ++j)
        async_copy16(&Ks[lrow * LD + lhalf + j * 8], gk + j * 8);
      // V block: batched loads into registers, then transposed LDS stores
      const uint4* gv = (const uint4*)(qkv + base + 2 * CC);
      uint4 vv[4];
#pragma unroll
      for (int j = 0; j < 4; ++j) vv[j] = gv[j];
#pragma unroll
      for (int j = 0; j < 4; ++j) {
        bf16 tmp[8];
        *(uint4*)tmp = vv[j];
#pragma unroll
        for (int e = 0; e < 8; ++e) Vt[(lhalf + j * 8 + e) * LD + lrow] = tmp[e];
      }
    }
    wait_async0();     // covers Q (first pass) and this block's K
    __syncthreads();

    // S = Q K^T  (16 rows x 64 keys per wave): 8 WMMAs
    v8f s[4];
#pragma unroll
    for (int nt = 0; nt < 4; ++nt) s[nt] = vzero;
#pragma unroll
    for (int ks = 0; ks < 64; ks += 32) {
      v16bf aq = ld_frag(Qs, LD, wid * 16 + (lane & 15), ks, lane);
#pragma unroll
      for (int nt = 0; nt < 4; ++nt) {
        v16bf bk = ld_frag(Ks, LD, nt * 16 + (lane & 15), ks, lane);
        s[nt] = wmma_bf16(aq, bk, s[nt]);
      }
    }

    // causal mask + online softmax (per-row across 16-lane half-waves)
    const float scale = 0.125f;   // 1/sqrt(64)
#pragma unroll
    for (int r = 0; r < 8; ++r) {
      int grow = rowbase + r;
      float mx = -1e30f;
#pragma unroll
      for (int nt = 0; nt < 4; ++nt) {
        float v = s[nt][r] * scale;
        int gcol = kb * 64 + nt * 16 + colb0;
        if (gcol > grow) v = -1e30f;
        s[nt][r] = v;
        mx = fmaxf(mx, v);
      }
#pragma unroll
      for (int off = 8; off > 0; off >>= 1) mx = fmaxf(mx, __shfl_xor(mx, off, 16));
      float mnew = fmaxf(m_i[r], mx);
      float alpha = __expf(m_i[r] - mnew);
      float rs = 0.f;
#pragma unroll
      for (int nt = 0; nt < 4; ++nt) {
        float p = __expf(s[nt][r] - mnew);
        s[nt][r] = p;
        rs += p;
      }
#pragma unroll
      for (int off = 8; off > 0; off >>= 1) rs += __shfl_xor(rs, off, 16);
      l_i[r] = l_i[r] * alpha + rs;
      m_i[r] = mnew;
#pragma unroll
      for (int dt = 0; dt < 4; ++dt) o[dt][r] *= alpha;
    }

    // spill P (C-fragment layout) to LDS as [row][key] bf16
#pragma unroll
    for (int nt = 0; nt < 4; ++nt)
#pragma unroll
      for (int r = 0; r < 8; ++r)
        Ps[(wid * 16 + r + ((lane & 16) ? 8 : 0)) * LD + nt * 16 + colb0] =
            (bf16)s[nt][r];
    __syncthreads();

    // O += P V  (P: 16x64, V: 64x64): 8 WMMAs
#pragma unroll
    for (int ks = 0; ks < 64; ks += 32) {
      v16bf ap = ld_frag(Ps, LD, wid * 16 + (lane & 15), ks, lane);
#pragma unroll
      for (int dt = 0; dt < 4; ++dt) {
        v16bf bv = ld_frag(Vt, LD, dt * 16 + (lane & 15), ks, lane);
        o[dt] = wmma_bf16(ap, bv, o[dt]);
      }
    }
  }

  // normalize + write attn output bf16 [B*T, C]
#pragma unroll
  for (int dt = 0; dt < 4; ++dt)
#pragma unroll
    for (int r = 0; r < 8; ++r) {
      int grow = rowbase + r;
      int gcol = hh * HD + dt * 16 + colb0;
      attnb[(size_t)(bb * TT + grow) * CC + gcol] =
          (bf16)(o[dt][r] / fmaxf(l_i[r], 1e-20f));
    }
}

// ---------------------------------------------------------------------------
// Orchestration
// ---------------------------------------------------------------------------
extern "C" void kernel_launch(void* const* d_in, const int* in_sizes, int n_in,
                              void* d_out, int out_size, void* d_ws, size_t ws_size,
                              hipStream_t stream) {
  (void)in_sizes; (void)n_in; (void)out_size; (void)ws_size;
  const int*   x      = (const int*)  d_in[0];
  const float* W_emb  = (const float*)d_in[1];
  const float* W_pos  = (const float*)d_in[2];
  const float* ln1_g  = (const float*)d_in[3];
  const float* ln1_b  = (const float*)d_in[4];
  const float* qkv_W  = (const float*)d_in[5];
  const float* qkv_b  = (const float*)d_in[6];
  const float* out_W  = (const float*)d_in[7];
  const float* out_b  = (const float*)d_in[8];
  const float* ln2_g  = (const float*)d_in[9];
  const float* ln2_b  = (const float*)d_in[10];
  const float* ffn_W1 = (const float*)d_in[11];
  const float* ffn_b1 = (const float*)d_in[12];
  const float* ffn_W2 = (const float*)d_in[13];
  const float* ffn_b2 = (const float*)d_in[14];
  const float* lnf_g  = (const float*)d_in[15];
  const float* lnf_b  = (const float*)d_in[16];

  // workspace carve (~44 MB)
  char* p = (char*)d_ws;
  float* h    = (float*)p; p += (size_t)MM * CC * sizeof(float);
  bf16*  a    = (bf16*) p; p += (size_t)MM * CC * sizeof(bf16);
  bf16*  qkvb = (bf16*) p; p += (size_t)MM * 3 * CC * sizeof(bf16);
  bf16*  attb = (bf16*) p; p += (size_t)MM * CC * sizeof(bf16);
  bf16*  ffnb = (bf16*) p; p += (size_t)MM * 4 * CC * sizeof(bf16);

  const dim3 blk256(256), blk128(128);

  embed_kernel<<<MM, blk256, 0, stream>>>(x, W_emb, W_pos, h);

  for (int l = 0; l < LLAYERS; ++l) {
    ln_kernel<<<MM, blk256, 0, stream>>>(h, ln1_g + (size_t)l * CC,
                                         ln1_b + (size_t)l * CC, a);
    gemm_kernel<<<dim3(3 * CC / 64, MM / 128), blk256, 0, stream>>>(
        a, qkv_W + (size_t)l * CC * 3 * CC, qkv_b + (size_t)l * 3 * CC,
        nullptr, nullptr, qkvb, 3 * CC, CC, 0, 0);
    attn_kernel<<<dim3(TT / 64, BB * HH), blk128, 0, stream>>>(qkvb, attb);
    gemm_kernel<<<dim3(CC / 64, MM / 128), blk256, 0, stream>>>(
        attb, out_W + (size_t)l * CC * CC, out_b + (size_t)l * CC,
        h, h, nullptr, CC, CC, 0, 0);
    ln_kernel<<<MM, blk256, 0, stream>>>(h, ln2_g + (size_t)l * CC,
                                         ln2_b + (size_t)l * CC, a);
    gemm_kernel<<<dim3(4 * CC / 64, MM / 128), blk256, 0, stream>>>(
        a, ffn_W1 + (size_t)l * CC * 4 * CC, ffn_b1 + (size_t)l * 4 * CC,
        nullptr, nullptr, ffnb, 4 * CC, CC, 0, 1);
    gemm_kernel<<<dim3(CC / 64, MM / 128), blk256, 0, stream>>>(
        ffnb, ffn_W2 + (size_t)l * 4 * CC * CC, ffn_b2 + (size_t)l * CC,
        h, h, nullptr, CC, 4 * CC, 0, 0);
  }

  ln_kernel<<<MM, blk256, 0, stream>>>(h, lnf_g, lnf_b, a);
  // logits = lnf(h) @ W_emb^T  (transposed-B path)
  gemm_kernel<<<dim3(VV / 64, MM / 128), blk256, 0, stream>>>(
      a, W_emb, nullptr, nullptr, (float*)d_out, nullptr, VV, CC, 1, 0);
}
